// Kernel_55722905698800
// MI455X (gfx1250) — matrix operand order repaired and checked by reading, not compiled, not run on hardware
//
#include <hip/hip_runtime.h>
#include <hip/hip_bf16.h>
#include <math.h>

typedef __bf16 bf16;
typedef __attribute__((ext_vector_type(16))) __bf16 v16bf;
typedef __attribute__((ext_vector_type(8)))  __bf16 v8bf;
typedef __attribute__((ext_vector_type(8)))  float  v8f;

#define E_TOTAL 20000
#define EPB 16           // edges per block (GEMM M-tile)
#define NB  64           // num radial basis (GEMM1 K)
#define HID 128          // hidden (GEMM1 N, GEMM2 K)
#define NOUT 1536        // 6 paths * 16 * 16 (GEMM2 N)

// ---------------------------------------------------------------------------
// Kernel 0: convert W1 (64x128) and W2 (128x1536) to transposed bf16 [n][k]
// so WMMA B-fragments are contiguous in K.
// ---------------------------------------------------------------------------
__global__ void convert_weights_kernel(const float* __restrict__ W1,
                                       const float* __restrict__ W2,
                                       bf16* __restrict__ W1t,
                                       bf16* __restrict__ W2t) {
  const int tid = blockIdx.x * blockDim.x + threadIdx.x;
  const int stride = gridDim.x * blockDim.x;
  for (int i = tid; i < NB * HID; i += stride) {
    const int k = i / HID, n = i - k * HID;      // coalesced read over n
    W1t[n * NB + k] = (bf16)W1[i];
  }
  for (int i = tid; i < HID * NOUT; i += stride) {
    const int k = i / NOUT, n = i - k * NOUT;
    W2t[n * HID + k] = (bf16)W2[i];
  }
}

// ---------------------------------------------------------------------------
// Fused kernel: 16 edges / block, 256 threads = 8 wave32s.
//   stage0: radius, basis(bf16->LDS), per-edge Wigner contraction mats
//   stage1: h = silu(basis @ W1 + b1)      WMMA bf16, 16x128
//   stage2: Rw = h @ W2 + b2 in two 768-col halves (f32 staged in LDS);
//           K-outer / N-inner with 6 live accumulators so the A fragment
//           is loaded once per K-step and shared across 6 WMMAs.
//   stage3: assemble 64x64 output blocks   (L-shape then 48x48)
// ---------------------------------------------------------------------------
__global__ void __launch_bounds__(256)
fused_kernel(const float* __restrict__ r,
             const float* __restrict__ b1,
             const float* __restrict__ b2,
             const float* __restrict__ wl0,
             const float* __restrict__ wl1,
             const bf16*  __restrict__ W1t,
             const bf16*  __restrict__ W2t,
             float* __restrict__ out)
{
  __shared__ __align__(32) bf16  s_basis[EPB * NB];   //  2 KB
  __shared__ __align__(32) bf16  s_h[EPB * HID];      //  4 KB
  __shared__ float s_Rw[EPB * 768];                   // 48 KB (one N-half)
  __shared__ float s_ed[EPB * 32];                    //  2 KB per-edge data

  const int tid  = threadIdx.x;
  const int lane = tid & 31;
  const int wave = tid >> 5;
  const int lh   = lane >> 4;     // lane half (0/1)
  const int ll   = lane & 15;
  const long long ebase = (long long)blockIdx.x * EPB;

  // ---------------- stage 0: per-edge geometry + basis ----------------
  if (tid < EPB) {
    const long long e = ebase + tid;
    const float rx = r[e*3+0], ry = r[e*3+1], rz = r[e*3+2];
    const float rad = sqrtf(rx*rx + ry*ry + rz*rz);
    const float inv = 1.0f / fmaxf(rad, 1e-12f);
    const float x = rx*inv, y = ry*inv, z = rz*inv;
    const float s3  = 1.7320508075688772f;            // sqrt(3)
    const float Y1v[3] = { s3*y, s3*z, s3*x };        // e3nn (y,z,x) order
    const float s15 = 3.872983346207417f;             // sqrt(15)
    const float Y2v[5] = { s15*x*y, s15*y*z,
                           1.118033988749895f*(3.0f*z*z - 1.0f),
                           s15*z*x, 0.5f*s15*(x*x - y*y) };
    float* ed = &s_ed[tid * 32];
    // path1 (0,1,1): c1[a] = Y1[a]/8      (norm sqrt3/8 * 1/sqrt3)
    ed[0] = Y1v[0]*0.125f; ed[1] = Y1v[1]*0.125f; ed[2] = Y1v[2]*0.125f;
    // path2 (1,0,1): c2[b] = Y1[b]/sqrt(96)
    const float c2s = 0.10206207261596575f;
    ed[3] = Y1v[0]*c2s; ed[4] = Y1v[1]*c2s; ed[5] = Y1v[2]*c2s;
    // path4 (1,1,1): C4[a][b] = 1/(8*sqrt2) * sum_f eps[a,b,f] Y1[f]
    const float e4 = 0.08838834764831845f;
    ed[6+0] = 0.0f;        ed[6+1] =  e4*Y1v[2]; ed[6+2] = -e4*Y1v[1];
    ed[6+3] = -e4*Y1v[2];  ed[6+4] = 0.0f;       ed[6+5] =  e4*Y1v[0];
    ed[6+6] =  e4*Y1v[1];  ed[6+7] = -e4*Y1v[0]; ed[6+8] = 0.0f;
    // path5 (1,1,2): C5[a][b] = sqrt3/(8*sqrt5) * sum_f T112raw[a,b,f] Y2[f]
    const float s5  = 0.09682458365518543f;
    const float s2c = 0.7071067811865476f;
    const float i6  = 0.4082482904638631f;
    ed[15+0] = s5*(-i6*Y2v[2] - s2c*Y2v[4]);
    ed[15+1] = s5*( s2c*Y2v[1]);
    ed[15+2] = s5*( s2c*Y2v[0]);
    ed[15+3] = s5*( s2c*Y2v[1]);
    ed[15+4] = s5*( 2.0f*i6*Y2v[2]);
    ed[15+5] = s5*( s2c*Y2v[3]);
    ed[15+6] = s5*( s2c*Y2v[0]);
    ed[15+7] = s5*( s2c*Y2v[3]);
    ed[15+8] = s5*(-i6*Y2v[2] + s2c*Y2v[4]);
    ed[24] = (rad > 0.0f) ? 1.0f : 0.0f;
  }
  for (int i = tid; i < EPB * NB; i += 256) {
    const int e = i >> 6, k = i & 63;
    const long long ge = ebase + e;
    const float rx = r[ge*3+0], ry = r[ge*3+1], rz = r[ge*3+2];
    const float rad = sqrtf(rx*rx + ry*ry + rz*rz);
    const float tt = (rad - (3.0f/63.0f)*(float)k) * (64.0f/3.0f);
    s_basis[i] = (bf16)__expf(-tt*tt);
  }
  __syncthreads();

  // ---------------- stage 1: h = silu(basis @ W1 + b1) ----------------
  {
    const int nn = wave * 16 + ll;                 // hidden column 0..127
    v8f acc = {};
#pragma unroll
    for (int kt = 0; kt < 2; ++kt) {
      const int k0 = kt * 32;
      const bf16* ap = &s_basis[ll*NB + k0 + lh*16];
      v8bf alo = *(const v8bf*)ap;
      v8bf ahi = *(const v8bf*)(ap + 8);
      v16bf a;
#pragma unroll
      for (int i = 0; i < 8; ++i) { a[i] = alo[i]; a[8+i] = ahi[i]; }
      v16bf bfrag = *(const v16bf*)&W1t[nn*NB + k0 + lh*16];
      acc = __builtin_amdgcn_wmma_f32_16x16x32_bf16(false, a, false, bfrag,
                                                    (short)0, acc, false, false);
    }
    const float bias = b1[nn];
#pragma unroll
    for (int j = 0; j < 8; ++j) {
      const int m = j + lh * 8;                    // edge row
      const float v = acc[j] + bias;
      // fast silu: v * rcp(1 + exp(-v))  (v_rcp_f32, no IEEE div sequence)
      const float s = v * __builtin_amdgcn_rcpf(1.0f + __expf(-v));
      s_h[m*HID + nn] = (bf16)s;
    }
  }
  __syncthreads();

  // ---------------- stage 2/3: two N-halves of Rw + assembly ----------------
  for (int half = 0; half < 2; ++half) {
    {
      // 6 N-tiles per wave, all accumulators live; A loaded once per K-step.
      v8f acc[6];
#pragma unroll
      for (int t = 0; t < 6; ++t) acc[t] = (v8f){};
#pragma unroll
      for (int kt = 0; kt < 4; ++kt) {
        const int k0 = kt * 32;
        const bf16* ap = &s_h[ll*HID + k0 + lh*16];
        v8bf alo = *(const v8bf*)ap;
        v8bf ahi = *(const v8bf*)(ap + 8);
        v16bf a;
#pragma unroll
        for (int i = 0; i < 8; ++i) { a[i] = alo[i]; a[8+i] = ahi[i]; }
#pragma unroll
        for (int t = 0; t < 6; ++t) {
          const int ng = half * 768 + (wave + 8*t) * 16 + ll;
          v16bf bfrag = *(const v16bf*)&W2t[ng*HID + k0 + lh*16];
          acc[t] = __builtin_amdgcn_wmma_f32_16x16x32_bf16(false, a, false, bfrag,
                                                           (short)0, acc[t],
                                                           false, false);
        }
      }
#pragma unroll
      for (int t = 0; t < 6; ++t) {
        const int nl = (wave + 8*t) * 16 + ll;     // local col 0..767
        const float bias = b2[half * 768 + nl];
#pragma unroll
        for (int j = 0; j < 8; ++j) {
          const int m = j + lh * 8;
          s_Rw[m*768 + nl] = acc[t][j] + bias;
        }
      }
    }
    __syncthreads();

    if (half == 0) {
      // L-shape: rows 0..15 (all cols) + rows 16..63 cols 0..15 ; paths 0,1,2
      const float n0c = 0.17677669529663687f;      // 1/sqrt(32)
      for (int t = tid; t < EPB * 1792; t += 256) {
        const int e = t / 1792;
        const int u = t - e * 1792;
        int row, col;
        if (u < 1024) { row = u >> 6; col = u & 63; }
        else { const int u2 = u - 1024; row = 16 + (u2 >> 4); col = u2 & 15; }
        const float* ed = &s_ed[e * 32];
        const float* Rw = &s_Rw[e * 768];
        float v;
        if (ed[24] > 0.5f) {
          if (row < 16) {
            if (col < 16) {
              v = Rw[0*256 + row*16 + col] * n0c;
            } else {
              const int c  = col - 16;
              const int ii = (int)(((unsigned)c * 171u) >> 9);
              const int b  = c - 3*ii;
              v = Rw[2*256 + row*16 + ii] * ed[3 + b];
            }
          } else {
            const int rr = row - 16;
            const int o  = (int)(((unsigned)rr * 171u) >> 9);
            const int a  = rr - 3*o;
            v = Rw[1*256 + o*16 + col] * ed[a];
          }
        } else {
          v = (row < 16 && col < 16) ? wl0[row*16 + col] * 0.25f : 0.0f;
        }
        out[(ebase + e) * 4096 + row*64 + col] = v;
      }
    } else {
      // bottom-right 48x48 ; paths 3,4,5 at local offsets 0,256,512
      for (int t = tid; t < EPB * 2304; t += 256) {
        const int e  = t / 2304;
        const int u  = t - e * 2304;
        const int rr = u / 48;
        const int cc = u - rr * 48;
        const int o  = (int)(((unsigned)rr * 171u) >> 9);
        const int a  = rr - 3*o;
        const int ii = (int)(((unsigned)cc * 171u) >> 9);
        const int b  = cc - 3*ii;
        const float* ed = &s_ed[e * 32];
        const float* Rw = &s_Rw[e * 768];
        float v;
        if (ed[24] > 0.5f) {
          const int oi = o*16 + ii;
          v = Rw[256 + oi] * ed[6 + a*3 + b] + Rw[512 + oi] * ed[15 + a*3 + b];
          if (a == b) v += Rw[oi] * 0.125f;
        } else {
          v = (a == b) ? wl1[o*16 + ii] * 0.25f : 0.0f;
        }
        out[(ebase + e) * 4096 + (16 + rr)*64 + (16 + cc)] = v;
      }
    }
    __syncthreads();
  }
}

// ---------------------------------------------------------------------------
extern "C" void kernel_launch(void* const* d_in, const int* in_sizes, int n_in,
                              void* d_out, int out_size, void* d_ws, size_t ws_size,
                              hipStream_t stream) {
  const float* r   = (const float*)d_in[0];
  const float* W1  = (const float*)d_in[1];
  const float* b1  = (const float*)d_in[2];
  const float* W2  = (const float*)d_in[3];
  const float* b2  = (const float*)d_in[4];
  const float* wl0 = (const float*)d_in[5];
  const float* wl1 = (const float*)d_in[6];
  float* out = (float*)d_out;

  bf16* W1t = (bf16*)d_ws;                 // 64*128  bf16 = 16 KB
  bf16* W2t = W1t + NB * HID;              // 1536*128 bf16 = 384 KB

  convert_weights_kernel<<<64, 256, 0, stream>>>(W1, W2, W1t, W2t);
  fused_kernel<<<E_TOTAL / EPB, 256, 0, stream>>>(r, b1, b2, wl0, wl1,
                                                  W1t, W2t, out);
}
